// AttentionCostVolume_73873437491334
// MI455X (gfx1250) — matrix operand order repaired and checked by reading, not compiled, not run on hardware
//
#include <hip/hip_runtime.h>
#include <hip/hip_bf16.h>

typedef __attribute__((ext_vector_type(16))) _Float16 v16h;
typedef __attribute__((ext_vector_type(8)))  _Float16 v8h;
typedef __attribute__((ext_vector_type(8)))  float    v8f;
typedef __attribute__((ext_vector_type(4)))  int      v4i;

// ---------------------------------------------------------------------------
// Geometry constants
// ---------------------------------------------------------------------------
#define BS   4
#define C    256
#define H    64
#define W    64
#define SR   8
#define MO   17           // 2*SR+1
#define NC   289          // MO*MO
#define M1   144          // agg1 out channels
#define M2   49           // agg2 out channels
#define KC1  320          // NC padded to 10*32 (per-plane K for agg1)
#define K1   (9*KC1)      // 2880
#define KC2  160          // M1 padded to 5*32 (per-plane K for agg2)
#define K2   (9*KC2)      // 1440
#define HP   66           // H + 2 (3x3 halo)
#define WP   66
#define HP2  80           // H + 2*SR (cost-volume pad)
#define WP2  80
#define PCOLS 34          // LDS patch width: 32 pixels + 2 halo

#define AS1 __attribute__((address_space(1)))
#define AS3 __attribute__((address_space(3)))

#if defined(__AMDGCN__) && __has_builtin(__builtin_amdgcn_global_load_async_to_lds_b128)
#define ASYNC_LDS 1
#endif

// ---------------------------------------------------------------------------
// Async (or fallback) 16B-chunk copy: contiguous global slice -> LDS
// ---------------------------------------------------------------------------
__device__ inline void lds_fill_chunk(const _Float16* g, _Float16* l) {
#if defined(ASYNC_LDS)
  __builtin_amdgcn_global_load_async_to_lds_b128(
      (AS1 v4i*)(void*)g, (AS3 v4i*)(void*)l, 0, 0);
#else
  *(v8h*)l = *(const v8h*)g;
#endif
}

__device__ inline void lds_fill_wait() {
#if defined(ASYNC_LDS)
#if __has_builtin(__builtin_amdgcn_s_wait_asynccnt)
  __builtin_amdgcn_s_wait_asynccnt(0);
#else
  asm volatile("s_wait_asynccnt 0" ::: "memory");
#endif
#endif
  __syncthreads();
}

// ---------------------------------------------------------------------------
// A-matrix fragment load (16x32 f16):
//   lanes 0-15  hold row M = lane,    K = {0..7, 16..23}
//   lanes 16-31 hold row M = lane-16, K = {8..15, 24..31}
// ---------------------------------------------------------------------------
__device__ inline v16h load_A(const _Float16* __restrict__ row, int lane) {
  const int kh = (lane & 16) ? 8 : 0;
  v8h lo = *(const v8h*)(row + kh);
  v8h hi = *(const v8h*)(row + kh + 16);
  v16h r;
#pragma unroll
  for (int i = 0; i < 8; ++i) { r[i] = lo[i]; r[i + 8] = hi[i]; }
  return r;
}

// ---------------------------------------------------------------------------
// B-matrix fragment load (32x16 f16):
//   lanes 0-15  hold col N = lane,    K = {0..7, 16..23}
//   lanes 16-31 hold col N = lane-16, K = {8..15, 24..31}
// Works for global or LDS (generic) pointers.
// ---------------------------------------------------------------------------
__device__ inline v16h load_B(const _Float16* col, int lane) {
  const int kh = (lane & 16) ? 8 : 0;
  v8h lo = *(const v8h*)(col + kh);
  v8h hi = *(const v8h*)(col + kh + 16);
  v16h r;
#pragma unroll
  for (int i = 0; i < 8; ++i) { r[i] = lo[i]; r[i + 8] = hi[i]; }
  return r;
}

// ---------------------------------------------------------------------------
// K1a: L2-normalize f1 over channels, convert to f16, layout [b][y][x][c]
// One wave per pixel (wave32).
// ---------------------------------------------------------------------------
__global__ __launch_bounds__(256) void k_normalize_f1(
    const float* __restrict__ f1, _Float16* __restrict__ f1n) {
  const int wg   = blockIdx.x * 8 + (threadIdx.x >> 5);   // pixel id, 0..16383
  const int lane = threadIdx.x & 31;
  const int x = wg & 63, y = (wg >> 6) & 63, b = wg >> 12;
  const float* src = f1 + (size_t)b * C * H * W + (size_t)y * W + x;
  float vals[8];
  float ss = 0.f;
#pragma unroll
  for (int i = 0; i < 8; ++i) {
    float v = src[(size_t)(lane + 32 * i) * (H * W)];
    vals[i] = v;
    ss += v * v;
  }
#pragma unroll
  for (int off = 16; off; off >>= 1) ss += __shfl_xor(ss, off, 32);
  const float inv = 1.0f / fmaxf(sqrtf(ss), 1e-12f);
  _Float16* dst = f1n + (size_t)wg * C;
#pragma unroll
  for (int i = 0; i < 8; ++i) dst[lane + 32 * i] = (_Float16)(vals[i] * inv);
}

// ---------------------------------------------------------------------------
// K1b: pad f2 spatially by SR and convert to f16, layout [b][y2][x2][c]
// ---------------------------------------------------------------------------
__global__ __launch_bounds__(256) void k_pad_f2(
    const float* __restrict__ f2, _Float16* __restrict__ f2h) {
  const size_t total = (size_t)BS * HP2 * WP2 * C;
  size_t idx = (size_t)blockIdx.x * 256 + threadIdx.x;
  if (idx >= total) return;
  const int c  = (int)(idx & (C - 1));
  const int x2 = (int)((idx >> 8) % WP2);
  const int y2 = (int)(((idx >> 8) / WP2) % HP2);
  const int b  = (int)(idx / ((size_t)HP2 * WP2 * C));
  const int x = x2 - SR, y = y2 - SR;
  float v = 0.f;
  if (x >= 0 && x < W && y >= 0 && y < H)
    v = f2[(((size_t)b * C + c) * H + y) * W + x];
  f2h[idx] = (_Float16)v;
}

// ---------------------------------------------------------------------------
// K2: cost volume via banded Gram WMMA.
// Block = (b, y, dy); 4 waves, wave w owns M-tile w (x1 in [16w,16w+16)),
// computes N-tiles {w, w+1} of G = F1row(64x256) x F2row(256x80),
// then scatters the dx-band with mean scale + leaky_relu(0.1).
// ---------------------------------------------------------------------------
__global__ __launch_bounds__(128) void k_cost_volume_wmma(
    const _Float16* __restrict__ f1n, const _Float16* __restrict__ f2h,
    float* __restrict__ mv) {
  const int blk = blockIdx.x;          // b*H*MO + y*MO + dy
  const int dy  = blk % MO;
  const int y   = (blk / MO) % H;
  const int b   = blk / (MO * H);
  const int lane  = threadIdx.x & 31;
  const int mtile = threadIdx.x >> 5;  // 0..3
  const int r16   = lane & 15;

  const _Float16* Abase =
      f1n + ((size_t)(b * H + y) * W + mtile * 16 + r16) * C;
  const int y2 = y + dy;
  const _Float16* B0base =
      f2h + ((size_t)(b * HP2 + y2) * WP2 + mtile * 16 + r16) * C;
  const _Float16* B1base = B0base + (size_t)16 * C;

  v8f acc[2] = {};
#pragma unroll
  for (int kk = 0; kk < 8; ++kk) {
    v16h a  = load_A(Abase  + kk * 32, lane);
    v16h b0 = load_B(B0base + kk * 32, lane);
    v16h b1 = load_B(B1base + kk * 32, lane);
    acc[0] = __builtin_amdgcn_wmma_f32_16x16x32_f16(
        false, a, false, b0, (short)0, acc[0], false, false);
    acc[1] = __builtin_amdgcn_wmma_f32_16x16x32_f16(
        false, a, false, b1, (short)0, acc[1], false, false);
  }

  const int mhi = (lane & 16) ? 8 : 0;
#pragma unroll
  for (int t = 0; t < 2; ++t) {
    const int n_global = (mtile + t) * 16 + r16;
#pragma unroll
    for (int r = 0; r < 8; ++r) {
      const int m_global = mtile * 16 + r + mhi;
      const int dx = n_global - m_global;
      if (dx >= 0 && dx < MO) {
        float v = acc[t][r] * (1.0f / 256.0f);
        v = v > 0.f ? v : 0.1f * v;                 // leaky_relu(0.1)
        const int o = dy * MO + dx;
        mv[(((size_t)b * NC + o) * H + y) * W + m_global] = v;
      }
    }
  }
}

// ---------------------------------------------------------------------------
// K3: depthwise 7x7 attention conv + gating; writes att_vol as f16 into a
// spatially padded, channel-contiguous buffer [b][HP][WP][KC1] (pad = 0).
// ---------------------------------------------------------------------------
__global__ __launch_bounds__(256) void k_dwconv_attvol(
    const float* __restrict__ mv, const float* __restrict__ attw,
    const float* __restrict__ attb, _Float16* __restrict__ avh) {
  const size_t total = (size_t)BS * HP * WP * KC1;
  size_t idx = (size_t)blockIdx.x * 256 + threadIdx.x;
  if (idx >= total) return;
  const int c  = (int)(idx % KC1);
  const int xx = (int)((idx / KC1) % WP);
  const int yy = (int)((idx / ((size_t)KC1 * WP)) % HP);
  const int b  = (int)(idx / ((size_t)KC1 * WP * HP));
  const int x = xx - 1, y = yy - 1;
  _Float16 outv = (_Float16)0.f;
  if (c < NC && x >= 0 && x < W && y >= 0 && y < H) {
    const float* plane = mv + ((size_t)b * NC + c) * (H * W);
    const float* w = attw + (size_t)c * 49;
    float acc = attb[c];
#pragma unroll
    for (int ky = 0; ky < 7; ++ky) {
      const int yk = y + ky - 3;
      if (yk < 0 || yk >= H) continue;
#pragma unroll
      for (int kx = 0; kx < 7; ++kx) {
        const int xk = x + kx - 3;
        if (xk < 0 || xk >= W) continue;
        acc += w[ky * 7 + kx] * plane[yk * W + xk];
      }
    }
    outv = (_Float16)(plane[y * W + x] * acc);    // gate
  }
  avh[idx] = outv;
}

// ---------------------------------------------------------------------------
// Weight packing: agg1 [144][289][3][3] f32 -> A1 [144][9*320] f16 (zero pad)
// ---------------------------------------------------------------------------
__global__ __launch_bounds__(256) void k_prep_w1(
    const float* __restrict__ w, _Float16* __restrict__ A1h) {
  const size_t total = (size_t)M1 * K1;
  size_t idx = (size_t)blockIdx.x * 256 + threadIdx.x;
  if (idx >= total) return;
  const int k  = (int)(idx % K1);
  const int oc = (int)(idx / K1);
  const int plane = k / KC1, c = k % KC1;
  const int ky = plane / 3, kx = plane % 3;
  float v = (c < NC) ? w[(((size_t)oc * NC + c) * 3 + ky) * 3 + kx] : 0.f;
  A1h[idx] = (_Float16)v;
}

// agg2 [49][144][3][3] f32 -> A2 [64][9*160] f16 (zero pad M and C)
__global__ __launch_bounds__(256) void k_prep_w2(
    const float* __restrict__ w, _Float16* __restrict__ A2h) {
  const size_t total = (size_t)64 * K2;
  size_t idx = (size_t)blockIdx.x * 256 + threadIdx.x;
  if (idx >= total) return;
  const int k  = (int)(idx % K2);
  const int oc = (int)(idx / K2);
  const int plane = k / KC2, c = k % KC2;
  const int ky = plane / 3, kx = plane % 3;
  float v = (oc < M2 && c < M1)
                ? w[(((size_t)oc * M1 + c) * 3 + ky) * 3 + kx] : 0.f;
  A2h[idx] = (_Float16)v;
}

__global__ __launch_bounds__(256) void k_zero_u64(
    unsigned long long* __restrict__ p, size_t n) {
  size_t i = (size_t)blockIdx.x * 256 + threadIdx.x;
  if (i < n) p[i] = 0ull;
}

// ---------------------------------------------------------------------------
// K4: agg1 implicit-GEMM WMMA with async-LDS B staging.
// Block = (b, y, xhalf): 32 output pixels of one row. 9 waves = 9 M-tiles,
// each wave accumulates 2 N-tiles. The 3x34x320 f16 im2col patch (65 KB) is
// staged once into LDS (GLOBAL_LOAD_ASYNC_TO_LDS_B128 when available), then
// all waves read B fragments via ds_load — removing the 9x redundant global
// B traffic. K = 9 planes x 320.
// ---------------------------------------------------------------------------
__global__ __launch_bounds__(288) void k_agg1_wmma(
    const _Float16* __restrict__ A1h, const _Float16* __restrict__ avh,
    const float* __restrict__ bias, _Float16* __restrict__ h1h) {
  __shared__ _Float16 sB[3 * PCOLS * KC1];          // 65,280 bytes

  const int blk = blockIdx.x;            // b*H*2 + y*2 + xh
  const int xh = blk & 1;
  const int y  = (blk >> 1) & 63;
  const int b  = blk >> 7;
  const int lane  = threadIdx.x & 31;
  const int mtile = threadIdx.x >> 5;    // 0..8
  const int r16   = lane & 15;
  const int x0g = xh * 32;               // first output pixel of this block

  // Stage B patch: rows y..y+2 (padded coords), cols x0g..x0g+33, 320 ch.
  {
    const _Float16* gpatch =
        avh + ((size_t)(b * HP + y) * WP + x0g) * KC1;
    const int row_chunks = PCOLS * KC1 / 8;          // 1360 x 16B per row
    for (int i = threadIdx.x; i < 3 * row_chunks; i += 288) {
      const int row = i / row_chunks;
      const int c16 = i - row * row_chunks;
      lds_fill_chunk(gpatch + (size_t)row * WP * KC1 + c16 * 8,
                     sB + row * (PCOLS * KC1) + c16 * 8);
    }
  }
  lds_fill_wait();

  const _Float16* Arow = A1h + (size_t)(mtile * 16 + r16) * K1;
  v8f acc[2] = {};
#pragma unroll
  for (int ky = 0; ky < 3; ++ky) {
#pragma unroll
    for (int kx = 0; kx < 3; ++kx) {
      const _Float16* Ap = Arow + (ky * 3 + kx) * KC1;
      __builtin_prefetch(Ap + KC1, 0, 1);            // global_prefetch_b8
#pragma unroll
      for (int kk = 0; kk < 10; ++kk) {
        v16h a = load_A(Ap + kk * 32, lane);
#pragma unroll
        for (int nt = 0; nt < 2; ++nt) {
          const _Float16* Bcol =
              sB + (ky * PCOLS + nt * 16 + r16 + kx) * KC1 + kk * 32;
          v16h bm = load_B(Bcol, lane);
          acc[nt] = __builtin_amdgcn_wmma_f32_16x16x32_f16(
              false, a, false, bm, (short)0, acc[nt], false, false);
        }
      }
    }
  }

  const int mhi = (lane & 16) ? 8 : 0;
#pragma unroll
  for (int nt = 0; nt < 2; ++nt) {
    const int xun = x0g + nt * 16 + r16;             // output x
#pragma unroll
    for (int r = 0; r < 8; ++r) {
      const int oc = mtile * 16 + r + mhi;
      float v = acc[nt][r] + bias[oc];
      v = v > 0.f ? v : 0.f;                         // relu
      h1h[((size_t)(b * HP + y + 1) * WP + (xun + 1)) * KC2 + oc] =
          (_Float16)v;
    }
  }
}

// ---------------------------------------------------------------------------
// K5: agg2 implicit-GEMM WMMA, same LDS staging scheme. M=49 padded to 64
// (4 waves), 2 N-tiles per wave, K = 9 x 160. Output -> relu -> fp32 d_out.
// ---------------------------------------------------------------------------
__global__ __launch_bounds__(128) void k_agg2_wmma(
    const _Float16* __restrict__ A2h, const _Float16* __restrict__ h1h,
    const float* __restrict__ bias, float* __restrict__ out) {
  __shared__ _Float16 sB[3 * PCOLS * KC2];          // 32,640 bytes

  const int blk = blockIdx.x;
  const int xh = blk & 1;
  const int y  = (blk >> 1) & 63;
  const int b  = blk >> 7;
  const int lane  = threadIdx.x & 31;
  const int mtile = threadIdx.x >> 5;    // 0..3
  const int r16   = lane & 15;
  const int x0g = xh * 32;

  {
    const _Float16* gpatch =
        h1h + ((size_t)(b * HP + y) * WP + x0g) * KC2;
    const int row_chunks = PCOLS * KC2 / 8;          // 680 x 16B per row
    for (int i = threadIdx.x; i < 3 * row_chunks; i += 128) {
      const int row = i / row_chunks;
      const int c16 = i - row * row_chunks;
      lds_fill_chunk(gpatch + (size_t)row * WP * KC2 + c16 * 8,
                     sB + row * (PCOLS * KC2) + c16 * 8);
    }
  }
  lds_fill_wait();

  const _Float16* Arow = A2h + (size_t)(mtile * 16 + r16) * K2;
  v8f acc[2] = {};
#pragma unroll
  for (int ky = 0; ky < 3; ++ky) {
#pragma unroll
    for (int kx = 0; kx < 3; ++kx) {
      const _Float16* Ap = Arow + (ky * 3 + kx) * KC2;
      __builtin_prefetch(Ap + KC2, 0, 1);
#pragma unroll
      for (int kk = 0; kk < 5; ++kk) {
        v16h a = load_A(Ap + kk * 32, lane);
#pragma unroll
        for (int nt = 0; nt < 2; ++nt) {
          const _Float16* Bcol =
              sB + (ky * PCOLS + nt * 16 + r16 + kx) * KC2 + kk * 32;
          v16h bm = load_B(Bcol, lane);
          acc[nt] = __builtin_amdgcn_wmma_f32_16x16x32_f16(
              false, a, false, bm, (short)0, acc[nt], false, false);
        }
      }
    }
  }

  const int mhi = (lane & 16) ? 8 : 0;
#pragma unroll
  for (int nt = 0; nt < 2; ++nt) {
    const int xun = x0g + nt * 16 + r16;
#pragma unroll
    for (int r = 0; r < 8; ++r) {
      const int oc = mtile * 16 + r + mhi;
      if (oc < M2) {
        float v = acc[nt][r] + bias[oc];
        v = v > 0.f ? v : 0.f;
        out[(((size_t)b * M2 + oc) * H + y) * W + xun] = v;
      }
    }
  }
}

// ---------------------------------------------------------------------------
// Host launcher
// ---------------------------------------------------------------------------
extern "C" void kernel_launch(void* const* d_in, const int* in_sizes, int n_in,
                              void* d_out, int out_size, void* d_ws,
                              size_t ws_size, hipStream_t stream) {
  (void)in_sizes; (void)n_in; (void)out_size; (void)ws_size;
  const float* f1     = (const float*)d_in[0];
  const float* f2     = (const float*)d_in[1];
  const float* att_w  = (const float*)d_in[2];
  const float* att_b  = (const float*)d_in[3];
  const float* agg1_w = (const float*)d_in[4];
  const float* agg1_b = (const float*)d_in[5];
  const float* agg2_w = (const float*)d_in[6];
  const float* agg2_b = (const float*)d_in[7];
  float* out = (float*)d_out;

  char* ws = (char*)d_ws;
  size_t off = 0;
  auto carve = [&](size_t bytes) -> char* {
    char* p = ws + off;
    off += (bytes + 255) & ~(size_t)255;
    return p;
  };
  _Float16* f1n = (_Float16*)carve((size_t)BS * H * W * C * 2);       //  8.4 MB
  _Float16* f2h = (_Float16*)carve((size_t)BS * HP2 * WP2 * C * 2);   // 13.1 MB
  float*    mv  = (float*)   carve((size_t)BS * NC * H * W * 4);      // 18.9 MB
  _Float16* avh = (_Float16*)carve((size_t)BS * HP * WP * KC1 * 2);   // 11.2 MB
  _Float16* h1h = (_Float16*)carve((size_t)BS * HP * WP * KC2 * 2);   //  5.6 MB
  _Float16* A1h = (_Float16*)carve((size_t)M1 * K1 * 2);              //  0.8 MB
  _Float16* A2h = (_Float16*)carve((size_t)64 * K2 * 2);              //  0.2 MB

  // Stage 0: packing / padding / normalization
  k_normalize_f1<<<(BS * H * W) / 8, 256, 0, stream>>>(f1, f1n);
  {
    const size_t n = (size_t)BS * HP2 * WP2 * C;
    k_pad_f2<<<(unsigned)((n + 255) / 256), 256, 0, stream>>>(f2, f2h);
  }
  {
    const size_t n = (size_t)M1 * K1;
    k_prep_w1<<<(unsigned)((n + 255) / 256), 256, 0, stream>>>(agg1_w, A1h);
  }
  {
    const size_t n = (size_t)64 * K2;
    k_prep_w2<<<(unsigned)((n + 255) / 256), 256, 0, stream>>>(agg2_w, A2h);
  }
  {
    const size_t n = (size_t)BS * HP * WP * KC2 * 2 / 8;   // u64 elements
    k_zero_u64<<<(unsigned)((n + 255) / 256), 256, 0, stream>>>(
        (unsigned long long*)h1h, n);
  }

  // Stage 1: cost volume (banded Gram, WMMA f16)
  k_cost_volume_wmma<<<BS * H * MO, 128, 0, stream>>>(f1n, f2h, mv);

  // Stage 2: depthwise 7x7 attention + gating -> padded f16 activations
  {
    const size_t n = (size_t)BS * HP * WP * KC1;
    k_dwconv_attvol<<<(unsigned)((n + 255) / 256), 256, 0, stream>>>(
        mv, att_w, att_b, avh);
  }

  // Stage 3: agg1 289->144 implicit GEMM (WMMA + async-LDS staging)
  k_agg1_wmma<<<BS * H * 2, 288, 0, stream>>>(A1h, avh, agg1_b, h1h);

  // Stage 4: agg2 144->49 implicit GEMM (WMMA + async-LDS staging) -> d_out
  k_agg2_wmma<<<BS * H * 2, 128, 0, stream>>>(A2h, h1h, agg2_b, out);
}
